// GraphProcessor_71820443124125
// MI455X (gfx1250) — matrix operand order repaired and checked by reading, not compiled, not run on hardware
//
#include <hip/hip_runtime.h>

// ---------------------------------------------------------------------------
// Types for gfx1250 WMMA
// ---------------------------------------------------------------------------
typedef __attribute__((ext_vector_type(16))) _Float16 v16h;
typedef __attribute__((ext_vector_type(8)))  _Float16 v8h;
typedef __attribute__((ext_vector_type(8)))  float    v8f;

#define LRELU_SLOPE 0.2f

// Sign-aware monotone float atomic-max (works for all non-NaN values).
__device__ __forceinline__ void atomicMaxF32(float* addr, float v) {
  if (v >= 0.0f) atomicMax((int*)addr, __float_as_int(v));
  else           atomicMin((unsigned int*)addr, __float_as_uint(v));
}

// ---------------------------------------------------------------------------
// Elementwise helpers
// ---------------------------------------------------------------------------
__global__ __launch_bounds__(256)
void fill_f32(float* __restrict__ p, int n, float v) {
  int i = blockIdx.x * 256 + threadIdx.x;
  if (i < n) p[i] = v;
}

__global__ __launch_bounds__(256)
void f32_to_f16(const float* __restrict__ in, _Float16* __restrict__ out, int n) {
  int i = blockIdx.x * 256 + threadIdx.x;
  if (i < n) out[i] = (_Float16)in[i];
}

// W [K,Nc] row-major f32  ->  WT [Nc,K] row-major f16 (i.e. column-major W)
__global__ __launch_bounds__(256)
void transpose_to_f16(const float* __restrict__ w, _Float16* __restrict__ wt,
                      int K, int Nc) {
  int i = blockIdx.x * 256 + threadIdx.x;
  if (i >= K * Nc) return;
  int n = i / K, k = i % K;
  wt[i] = (_Float16)w[k * Nc + n];
}

// ---------------------------------------------------------------------------
// f16 WMMA GEMM:  C[M,Ncols] = A[M,K] * B[K,Ncols],  B given transposed (BT[Ncols,K])
// One wave computes one 16x16 C tile via v_wmma_f32_16x16x32_f16 over K/32 steps.
// Operand layouts per CDNA5 ISA 7.12.2.
// ---------------------------------------------------------------------------
__global__ __launch_bounds__(128)
void gemm_f16_wmma(const _Float16* __restrict__ A,
                   const _Float16* __restrict__ BT,
                   float* __restrict__ C,
                   int M, int Ncols, int K) {
  const int wave    = threadIdx.x >> 5;
  const int lane    = threadIdx.x & 31;
  const int half_id = lane >> 4;     // 0: lanes 0-15, 1: lanes 16-31
  const int r       = lane & 15;
  const int tm      = blockIdx.x;
  const int tn      = blockIdx.y * 4 + wave;
  const int tilesN  = Ncols >> 4;
  if (tn >= tilesN) return;          // wave-uniform: EXEC stays all-1 for WMMA

  int rowA = tm * 16 + r;
  if (rowA >= M) rowA = M - 1;       // clamp loads; stores are guarded below
  const int colB = tn * 16 + r;

  // A lane view: row rowA, halves at K = k0 + half_id*8 + {0..7} and +16
  const _Float16* aPtr = A  + (size_t)rowA * K + half_id * 8;
  // B lane view: column colB, halves at K = k0 + half_id*8 + {0..7} and +16
  const _Float16* bPtr = BT + (size_t)colB * K + half_id * 8;

  v8f acc = {};
  for (int k0 = 0; k0 < K; k0 += 32) {
    v8h a0 = *(const v8h*)(aPtr + k0);
    v8h a1 = *(const v8h*)(aPtr + k0 + 16);
    v16h av, bv;
    v8h b0 = *(const v8h*)(bPtr + k0);
    v8h b1 = *(const v8h*)(bPtr + k0 + 16);
#pragma unroll
    for (int i = 0; i < 8; ++i) { av[i] = a0[i]; av[8 + i] = a1[i]; bv[i] = b0[i]; bv[8 + i] = b1[i]; }
    acc = __builtin_amdgcn_wmma_f32_16x16x32_f16(false, av, false, bv,
                                                 (short)0, acc, false, false);
  }

  // C/D layout: lane half 0 -> M = tm*16 + v, half 1 -> M = tm*16 + 8 + v; N = r
  const int rowBase = tm * 16 + half_id * 8;
  const int col     = tn * 16 + r;
#pragma unroll
  for (int v = 0; v < 8; ++v) {
    int row = rowBase + v;
    if (row < M) C[(size_t)row * Ncols + col] = acc[v];
  }
}

// ---------------------------------------------------------------------------
// GAT edge pipeline
// ---------------------------------------------------------------------------
template <int H, int CH>
__global__ __launch_bounds__(256)
void node_coeffs(const float* __restrict__ h, const float* __restrict__ a_src,
                 const float* __restrict__ a_dst,
                 float* __restrict__ asrc, float* __restrict__ adst, int N) {
  int idx = blockIdx.x * 256 + threadIdx.x;
  if (idx >= N * H) return;
  int n = idx / H, hd = idx % H;
  const float* row = h + (size_t)n * (H * CH) + hd * CH;
  float s = 0.0f, d = 0.0f;
#pragma unroll 4
  for (int c = 0; c < CH; ++c) {
    float v = row[c];
    s += v * a_src[hd * CH + c];
    d += v * a_dst[hd * CH + c];
  }
  asrc[idx] = s;
  adst[idx] = d;
}

__device__ __forceinline__ void edge_endpoints(const int* ei, int E_in, int e,
                                               int& s, int& d) {
  if (e < E_in) { s = ei[e]; d = ei[E_in + e]; }
  else          { s = e - E_in; d = s; }          // appended self-loops
}

template <int H>
__global__ __launch_bounds__(256)
void edge_attn_max(const int* __restrict__ ei, int E_in, int E,
                   const float* __restrict__ asrc, const float* __restrict__ adst,
                   float* __restrict__ m) {
  int e = blockIdx.x * 256 + threadIdx.x;
  if (e >= E) return;
  int s, d; edge_endpoints(ei, E_in, e, s, d);
#pragma unroll
  for (int h = 0; h < H; ++h) {
    float v = asrc[s * H + h] + adst[d * H + h];
    v = v > 0.0f ? v : LRELU_SLOPE * v;
    atomicMaxF32(&m[d * H + h], v);
  }
}

template <int H>
__global__ __launch_bounds__(256)
void edge_attn_exp(const int* __restrict__ ei, int E_in, int E,
                   const float* __restrict__ asrc, const float* __restrict__ adst,
                   const float* __restrict__ m,
                   float* __restrict__ ex, float* __restrict__ den) {
  int e = blockIdx.x * 256 + threadIdx.x;
  if (e >= E) return;
  int s, d; edge_endpoints(ei, E_in, e, s, d);
#pragma unroll
  for (int h = 0; h < H; ++h) {
    float v = asrc[s * H + h] + adst[d * H + h];
    v = v > 0.0f ? v : LRELU_SLOPE * v;
    float exv = expf(v - m[d * H + h]);
    ex[e * H + h] = exv;
    atomicAdd(&den[d * H + h], exv);
  }
}

template <int H, int CH>
__global__ __launch_bounds__(256)
void edge_msg(const int* __restrict__ ei, int E_in, int E,
              const float* __restrict__ hfeat, const float* __restrict__ ex,
              const float* __restrict__ den, float* __restrict__ agg) {
  const int HC = H * CH;
  long long idx = (long long)blockIdx.x * 256 + threadIdx.x;
  if (idx >= (long long)E * HC) return;
  int e   = (int)(idx / HC);
  int col = (int)(idx % HC);
  int hd  = col / CH;
  int s, d; edge_endpoints(ei, E_in, e, s, d);
  float alpha = ex[e * H + hd] / den[d * H + hd];
  atomicAdd(&agg[(size_t)d * HC + col], alpha * hfeat[(size_t)s * HC + col]);
}

// relu(agg + b) -> f16 (input to next layer's WMMA GEMM)
__global__ __launch_bounds__(256)
void finalize_relu_f16(const float* __restrict__ agg, const float* __restrict__ b,
                       _Float16* __restrict__ out, int N, int C) {
  int i = blockIdx.x * 256 + threadIdx.x;
  if (i >= N * C) return;
  float v = agg[i] + b[i % C];
  out[i] = (_Float16)(v > 0.0f ? v : 0.0f);
}

// relu(agg + b) then scatter-add into per-graph sums
__global__ __launch_bounds__(256)
void finalize_pool(const float* __restrict__ agg, const float* __restrict__ b,
                   const int* __restrict__ batch, float* __restrict__ gsum,
                   int N, int C) {
  int i = blockIdx.x * 256 + threadIdx.x;
  if (i >= N * C) return;
  int n = i / C, c = i % C;
  float v = agg[i] + b[c];
  v = v > 0.0f ? v : 0.0f;
  atomicAdd(&gsum[batch[n] * C + c], v);
}

__global__ __launch_bounds__(256)
void count_nodes(const int* __restrict__ batch, float* __restrict__ gcnt, int N) {
  int n = blockIdx.x * 256 + threadIdx.x;
  if (n < N) atomicAdd(&gcnt[batch[n]], 1.0f);
}

__global__ __launch_bounds__(256)
void pool_div(const float* __restrict__ gsum, const float* __restrict__ gcnt,
              float* __restrict__ pooled, int G, int C) {
  int i = blockIdx.x * 256 + threadIdx.x;
  if (i >= G * C) return;
  pooled[i] = gsum[i] / fmaxf(gcnt[i / C], 1.0f);
}

// small dense layer: out[g,o] = act( sum_c in[g,c]*w[c,o] + bias[o] )
__global__ __launch_bounds__(256)
void dense(const float* __restrict__ in, const float* __restrict__ w,
           const float* __restrict__ bias, float* __restrict__ out,
           int G, int IN, int OUT, int do_relu) {
  int i = blockIdx.x * 256 + threadIdx.x;
  if (i >= G * OUT) return;
  int g = i / OUT, o = i % OUT;
  float acc = bias[o];
  for (int c = 0; c < IN; ++c) acc += in[g * IN + c] * w[c * OUT + o];
  out[i] = do_relu ? fmaxf(acc, 0.0f) : acc;
}

// ---------------------------------------------------------------------------
// Host-side orchestration
// ---------------------------------------------------------------------------
static inline dim3 nblk(long long n) { return dim3((unsigned)((n + 255) / 256)); }

extern "C" void kernel_launch(void* const* d_in, const int* in_sizes, int n_in,
                              void* d_out, int out_size, void* d_ws, size_t ws_size,
                              hipStream_t stream) {
  const float* x      = (const float*)d_in[0];
  const int*   ei     = (const int*)  d_in[1];
  const int*   batch  = (const int*)  d_in[2];
  const float* W1     = (const float*)d_in[3];
  const float* a_src1 = (const float*)d_in[4];
  const float* a_dst1 = (const float*)d_in[5];
  const float* b1     = (const float*)d_in[6];
  const float* W2     = (const float*)d_in[7];
  const float* a_src2 = (const float*)d_in[8];
  const float* a_dst2 = (const float*)d_in[9];
  const float* b2     = (const float*)d_in[10];
  const float* fc1_w  = (const float*)d_in[11];
  const float* fc1_b  = (const float*)d_in[12];
  const float* fc2_w  = (const float*)d_in[13];
  const float* fc2_b  = (const float*)d_in[14];
  float* out = (float*)d_out;

  const int N    = in_sizes[0] / 128;   // 100000 nodes
  const int E_in = in_sizes[1] / 2;     // 1,600,000 edges
  const int E    = E_in + N;            // + self-loops
  const int G    = out_size / 96;       // 128 graphs

  // bump allocator over scratch
  char* p = (char*)d_ws;
  auto alloc = [&](size_t bytes) -> void* {
    void* r = (void*)p;
    p += (bytes + 255) & ~((size_t)255);
    return r;
  };

  _Float16* xh    = (_Float16*)alloc((size_t)N * 128 * sizeof(_Float16));
  _Float16* w1t   = (_Float16*)alloc((size_t)192 * 128 * sizeof(_Float16));
  float*    h1    = (float*)   alloc((size_t)N * 192 * sizeof(float));
  float*    as1   = (float*)   alloc((size_t)N * 4 * sizeof(float));
  float*    ad1   = (float*)   alloc((size_t)N * 4 * sizeof(float));
  float*    m1    = (float*)   alloc((size_t)N * 4 * sizeof(float));
  float*    den1  = (float*)   alloc((size_t)N * 4 * sizeof(float));
  float*    ex1   = (float*)   alloc((size_t)E * 4 * sizeof(float));
  float*    agg1  = (float*)   alloc((size_t)N * 192 * sizeof(float));
  _Float16* o1h   = (_Float16*)alloc((size_t)N * 192 * sizeof(_Float16));
  _Float16* w2t   = (_Float16*)alloc((size_t)96 * 192 * sizeof(_Float16));
  float*    h2    = (float*)   alloc((size_t)N * 96 * sizeof(float));
  float*    as2   = (float*)   alloc((size_t)N * sizeof(float));
  float*    ad2   = (float*)   alloc((size_t)N * sizeof(float));
  float*    m2    = (float*)   alloc((size_t)N * sizeof(float));
  float*    den2  = (float*)   alloc((size_t)N * sizeof(float));
  float*    ex2   = (float*)   alloc((size_t)E * sizeof(float));
  float*    agg2  = (float*)   alloc((size_t)N * 96 * sizeof(float));
  float*    gsum  = (float*)   alloc((size_t)G * 96 * sizeof(float));
  float*    gcnt  = (float*)   alloc((size_t)G * sizeof(float));
  float*    pooled= (float*)   alloc((size_t)G * 96 * sizeof(float));
  float*    z     = (float*)   alloc((size_t)G * 192 * sizeof(float));

  const int tilesM = (N + 15) / 16;

  // ---- Layer 1: h1 = x @ W1 (f16 WMMA, f32 accumulate) ----
  f32_to_f16<<<nblk((long long)N * 128), 256, 0, stream>>>(x, xh, N * 128);
  transpose_to_f16<<<nblk(192 * 128), 256, 0, stream>>>(W1, w1t, 128, 192);
  gemm_f16_wmma<<<dim3(tilesM, 3), 128, 0, stream>>>(xh, w1t, h1, N, 192, 128);

  node_coeffs<4, 48><<<nblk((long long)N * 4), 256, 0, stream>>>(h1, a_src1, a_dst1,
                                                                 as1, ad1, N);
  fill_f32<<<nblk((long long)N * 4), 256, 0, stream>>>(m1, N * 4, -INFINITY);
  fill_f32<<<nblk((long long)N * 4), 256, 0, stream>>>(den1, N * 4, 0.0f);
  fill_f32<<<nblk((long long)N * 192), 256, 0, stream>>>(agg1, N * 192, 0.0f);

  edge_attn_max<4><<<nblk(E), 256, 0, stream>>>(ei, E_in, E, as1, ad1, m1);
  edge_attn_exp<4><<<nblk(E), 256, 0, stream>>>(ei, E_in, E, as1, ad1, m1, ex1, den1);
  edge_msg<4, 48><<<nblk((long long)E * 192), 256, 0, stream>>>(ei, E_in, E, h1,
                                                                ex1, den1, agg1);
  finalize_relu_f16<<<nblk((long long)N * 192), 256, 0, stream>>>(agg1, b1, o1h,
                                                                  N, 192);

  // ---- Layer 2: h2 = relu(out1) @ W2 ----
  transpose_to_f16<<<nblk(96 * 192), 256, 0, stream>>>(W2, w2t, 192, 96);
  gemm_f16_wmma<<<dim3(tilesM, 2), 128, 0, stream>>>(o1h, w2t, h2, N, 96, 192);

  node_coeffs<1, 96><<<nblk(N), 256, 0, stream>>>(h2, a_src2, a_dst2, as2, ad2, N);
  fill_f32<<<nblk(N), 256, 0, stream>>>(m2, N, -INFINITY);
  fill_f32<<<nblk(N), 256, 0, stream>>>(den2, N, 0.0f);
  fill_f32<<<nblk((long long)N * 96), 256, 0, stream>>>(agg2, N * 96, 0.0f);
  fill_f32<<<nblk(G * 96), 256, 0, stream>>>(gsum, G * 96, 0.0f);
  fill_f32<<<nblk(G), 256, 0, stream>>>(gcnt, G, 0.0f);

  edge_attn_max<1><<<nblk(E), 256, 0, stream>>>(ei, E_in, E, as2, ad2, m2);
  edge_attn_exp<1><<<nblk(E), 256, 0, stream>>>(ei, E_in, E, as2, ad2, m2, ex2, den2);
  edge_msg<1, 96><<<nblk((long long)E * 96), 256, 0, stream>>>(ei, E_in, E, h2,
                                                               ex2, den2, agg2);

  // ---- Global mean pool + FC head ----
  finalize_pool<<<nblk((long long)N * 96), 256, 0, stream>>>(agg2, b2, batch, gsum,
                                                             N, 96);
  count_nodes<<<nblk(N), 256, 0, stream>>>(batch, gcnt, N);
  pool_div<<<nblk(G * 96), 256, 0, stream>>>(gsum, gcnt, pooled, G, 96);
  dense<<<nblk(G * 192), 256, 0, stream>>>(pooled, fc1_w, fc1_b, z, G, 96, 192, 1);
  dense<<<nblk(G * 96), 256, 0, stream>>>(z, fc2_w, fc2_b, out, G, 192, 96, 0);
}
